// HOPEProjection_84387517432345
// MI455X (gfx1250) — compile-verified, matrix operand order checked by reading, not run on hardware
//
#include <hip/hip_runtime.h>
#include <math.h>

// ---------------------------------------------------------------------------
// HOPE projection, fused for MI455X (gfx1250, wave32, WMMA f16 16x16x32).
//
//   LN(x) @ W1 -> GELU -> @ W2 -> tanh+1, with LayerNorm folded into the GEMM:
//     h_pre[b,n] = rstd[b] * ( (x @ W1')[b,n] - mu[b]*s1[n] ) + bias1'[n]
//   where W1' = gamma (row) * W1,  s1 = colsum(W1'),  bias1' = b1 + beta @ W1.
// ---------------------------------------------------------------------------

typedef __attribute__((ext_vector_type(16))) _Float16 v16h;
typedef __attribute__((ext_vector_type(8)))  _Float16 v8h;
typedef __attribute__((ext_vector_type(4)))  _Float16 v4h;
typedef __attribute__((ext_vector_type(8)))  float    v8f;
typedef __attribute__((ext_vector_type(4)))  float    v4f;

#define D_MODEL 2048
#define HIDDEN  512
#define NPARAM  96     // 3 * n_heads
#define NHEADS  32
#define MTILE   64     // rows per block
#define KT1     (D_MODEL / 32)   // 64 k-steps for GEMM1
#define KT2     (HIDDEN  / 32)   // 16 k-steps for GEMM2
#define NT2     (NPARAM  / 16)   // 6 n-tiles for GEMM2

#define XSTRIDE 2056   // halves per LDS x-row (2048 + 8 pad, keeps 16B align)
#define HSTRIDE 520    // halves per LDS h-row (512 + 8 pad)

#define X_BYTES    (MTILE * XSTRIDE * 2)                    // 263,168
#define SMEM_BYTES (X_BYTES + 512*4 + 512*4 + 64*4 + 64*4)  // 267,776

// Hardware transcendental tanh if the toolchain exposes it (CDNA5 TRANS op).
#if __has_builtin(__builtin_amdgcn_tanhf)
__device__ __forceinline__ float fast_tanh(float x) { return __builtin_amdgcn_tanhf(x); }
#elif __has_builtin(__builtin_amdgcn_tanh_f32)
__device__ __forceinline__ float fast_tanh(float x) { return __builtin_amdgcn_tanh_f32(x); }
#else
__device__ __forceinline__ float fast_tanh(float x) { return tanhf(x); }
#endif

// WGP-scope prefetch (scope 0 = pull into ALL cache levels, incl. WGP$).
// __builtin_prefetch can only reach SE/SYS scopes, which skip the WGP cache,
// so emit the CDNA5 instruction directly with the default (WGP) scope.
__device__ __forceinline__ void prefetch_wgp_8k(const void* p) {
    asm volatile("global_prefetch_b8 %0, off offset:8192" :: "v"(p));
}

// K index inside a 16-bit 16x32 WMMA fragment for (lane, elem e):
//   lanes 0-15 : e0..7 -> K 0..7,  e8..15 -> K 16..23
//   lanes 16-31: e0..7 -> K 8..15, e8..15 -> K 24..31
__device__ __forceinline__ int frag_k(int lane, int e) {
    return (e < 8 ? e : e + 8) + ((lane >= 16) ? 8 : 0);
}

// ---------------------- prep: pack W1' = gamma * W1 into B-fragments --------
__global__ __launch_bounds__(256) void pack_w1(const float* __restrict__ W1,
                                               const float* __restrict__ gamma,
                                               _Float16* __restrict__ B1pack) {
    int i    = blockIdx.x * 256 + threadIdx.x;    // over 2048*512 halves
    int e    = i & 15;
    int lane = (i >> 4) & 31;
    int frag = i >> 9;
    int kt   = frag % KT1;
    int nt   = frag / KT1;
    int n    = nt * 16 + (lane & 15);
    int kk   = kt * 32 + frag_k(lane, e);
    B1pack[i] = (_Float16)(gamma[kk] * W1[(size_t)kk * HIDDEN + n]);
}

// ---------------------- prep: pack W2 into B-fragments ----------------------
__global__ __launch_bounds__(256) void pack_w2(const float* __restrict__ W2,
                                               _Float16* __restrict__ B2pack) {
    int i    = blockIdx.x * 256 + threadIdx.x;    // over 512*96 halves
    int e    = i & 15;
    int lane = (i >> 4) & 31;
    int frag = i >> 9;
    int kt   = frag % KT2;
    int nt   = frag / KT2;
    int n    = nt * 16 + (lane & 15);
    int kk   = kt * 32 + frag_k(lane, e);
    B2pack[i] = (_Float16)W2[(size_t)kk * NPARAM + n];
}

// ---------------------- prep: s1[n], bias1'[n] ------------------------------
__global__ __launch_bounds__(256) void col_reduce(const float* __restrict__ W1,
                                                  const float* __restrict__ gamma,
                                                  const float* __restrict__ beta,
                                                  const float* __restrict__ b1,
                                                  float* __restrict__ s1,
                                                  float* __restrict__ b1p) {
    __shared__ float rs[256], rb[256];
    int n = blockIdx.x, t = threadIdx.x;
    float a = 0.f, b = 0.f;
    for (int k = t; k < D_MODEL; k += 256) {
        float w = W1[(size_t)k * HIDDEN + n];
        a += gamma[k] * w;
        b += beta[k] * w;
    }
    rs[t] = a; rb[t] = b;
    __syncthreads();
    for (int s = 128; s > 0; s >>= 1) {
        if (t < s) { rs[t] += rs[t + s]; rb[t] += rb[t + s]; }
        __syncthreads();
    }
    if (t == 0) { s1[n] = rs[0]; b1p[n] = rb[0] + b1[n]; }
}

// ---------------------- fused main kernel -----------------------------------
__global__ __launch_bounds__(512, 1)
void hope_main(const float* __restrict__ slow,
               const _Float16* __restrict__ B1pack,
               const _Float16* __restrict__ B2pack,
               const float* __restrict__ s1,
               const float* __restrict__ b1p,
               const float* __restrict__ b2,
               float* __restrict__ out,
               int Btot) {
    extern __shared__ __align__(16) char smem[];
    _Float16* xs     = (_Float16*)smem;                 // 64 x 2056 halves (reused as h)
    float*    psum   = (float*)(smem + X_BYTES);        // 512
    float*    psumsq = psum + 512;                      // 512
    float*    ldsMu  = psumsq + 512;                    // 64
    float*    ldsRs  = ldsMu + 64;                      // 64

    const int tid  = threadIdx.x;
    const int row0 = blockIdx.x * MTILE;
    const int wave = tid >> 5;
    const int lane = tid & 31;
    const int lh   = lane & 15;
    const int hiK  = (lane >= 16) ? 8 : 0;
    const int hiM  = (lane >= 16) ? 8 : 0;

    // ---- Phase 1: stage x rows as f16 into LDS (read slow_state once) ----
    #pragma unroll 4
    for (int it = 0; it < (MTILE * D_MODEL / 4) / 512; ++it) {  // 64 iters
        int c  = it * 512 + tid;
        int r  = c >> 9;          // 512 float4 per row
        int c4 = c & 511;
        v4f v = *(const v4f*)(slow + (size_t)(row0 + r) * D_MODEL + c4 * 4);
        v4h h = { (_Float16)v[0], (_Float16)v[1], (_Float16)v[2], (_Float16)v[3] };
        *(v4h*)(xs + (size_t)r * XSTRIDE + c4 * 4) = h;
    }
    __syncthreads();

    // ---- Phase 1b: per-row mean / rstd (8 partials per row) ----
    {
        int r = tid >> 3, part = tid & 7;
        const _Float16* rp = xs + (size_t)r * XSTRIDE + part * 256;
        float s = 0.f, sq = 0.f;
        for (int u = 0; u < 32; ++u) {
            v8h hv = *(const v8h*)(rp + u * 8);
            #pragma unroll
            for (int e = 0; e < 8; ++e) { float f = (float)hv[e]; s += f; sq += f * f; }
        }
        psum[tid] = s; psumsq[tid] = sq;
    }
    __syncthreads();
    if (tid < MTILE) {
        float s = 0.f, sq = 0.f;
        #pragma unroll
        for (int p = 0; p < 8; ++p) { s += psum[tid * 8 + p]; sq += psumsq[tid * 8 + p]; }
        float mu  = s * (1.0f / D_MODEL);
        float var = sq * (1.0f / D_MODEL) - mu * mu;
        ldsMu[tid] = mu;
        ldsRs[tid] = rsqrtf(var + 1e-5f);
    }
    __syncthreads();

    // ---- Phase 2: GEMM1, x(f16,LDS) @ W1'(f16 fragments, L2) ----
    // wave owns n-columns [wave*32, wave*32+32): 2 n-tiles x 4 m-tiles.
    v8f acc[4][2] = {};
    #pragma unroll 2
    for (int kt = 0; kt < KT1; ++kt) {
        int kb = kt * 32 + hiK;
        v16h a[4];
        #pragma unroll
        for (int mt = 0; mt < 4; ++mt) {
            const _Float16* ap = xs + (size_t)(mt * 16 + lh) * XSTRIDE + kb;
            v8h lo = *(const v8h*)ap;
            v8h hi = *(const v8h*)(ap + 16);
            a[mt] = __builtin_shufflevector(lo, hi, 0,1,2,3,4,5,6,7,8,9,10,11,12,13,14,15);
        }
        #pragma unroll
        for (int j = 0; j < 2; ++j) {
            int nt = wave * 2 + j;
            const _Float16* bp = B1pack + ((size_t)(nt * KT1 + kt) * 32 + lane) * 16;
            // WGP-scope prefetch 8 k-steps (8KB) ahead; overrunning the end of
            // B1pack just touches B2pack (same workspace) - harmless.
            prefetch_wgp_8k(bp);
            v16h b = *(const v16h*)bp;
            #pragma unroll
            for (int mt = 0; mt < 4; ++mt)
                acc[mt][j] = __builtin_amdgcn_wmma_f32_16x16x32_f16(
                    false, a[mt], false, b, (short)0, acc[mt][j], false, false);
        }
    }
    __syncthreads();   // everyone done reading xs before we overwrite it with h

    // ---- Phase 3: LN-correction + exact GELU, write h (f16) into LDS ----
    #pragma unroll
    for (int j = 0; j < 2; ++j) {
        int col = wave * 32 + j * 16 + lh;
        float s1v = s1[col];
        float b1v = b1p[col];
        #pragma unroll
        for (int mt = 0; mt < 4; ++mt) {
            #pragma unroll
            for (int i = 0; i < 8; ++i) {
                int row = mt * 16 + i + hiM;
                float pre = ldsRs[row] * (acc[mt][j][i] - ldsMu[row] * s1v) + b1v;
                float g = 0.5f * pre * (1.0f + erff(pre * 0.70710678118654752f));
                xs[(size_t)row * HSTRIDE + col] = (_Float16)g;
            }
        }
    }
    __syncthreads();

    // ---- Phase 4: GEMM2, h(f16,LDS) @ W2(f16 fragments) -> tanh+1 -> out ----
    // 24 tile jobs (4 m-tiles x 6 n-tiles) over 16 waves; wave-uniform loop,
    // so EXEC stays all-ones for every WMMA.
    for (int job = wave; job < 4 * NT2; job += 16) {
        int m2 = job & 3;
        int n2 = job >> 2;
        v8f c2 = {};
        for (int kt = 0; kt < KT2; ++kt) {
            int kb = kt * 32 + hiK;
            const _Float16* ap = xs + (size_t)(m2 * 16 + lh) * HSTRIDE + kb;
            v8h lo = *(const v8h*)ap;
            v8h hi = *(const v8h*)(ap + 16);
            v16h a2 = __builtin_shufflevector(lo, hi, 0,1,2,3,4,5,6,7,8,9,10,11,12,13,14,15);
            v16h b  = *(const v16h*)(B2pack + ((size_t)(n2 * KT2 + kt) * 32 + lane) * 16);
            c2 = __builtin_amdgcn_wmma_f32_16x16x32_f16(
                false, a2, false, b, (short)0, c2, false, false);
        }
        int col  = n2 * 16 + lh;          // 0..95 = 3*head + comp
        int head = col / 3;
        int comp = col % 3;
        float b2v = b2[col];
        #pragma unroll
        for (int i = 0; i < 8; ++i) {
            int grow = row0 + m2 * 16 + i + hiM;
            float val = fast_tanh(c2[i] + b2v) + 1.0f;
            out[(size_t)comp * Btot * NHEADS + (size_t)grow * NHEADS + head] = val;
        }
    }
}

// ---------------------------------------------------------------------------
extern "C" void kernel_launch(void* const* d_in, const int* in_sizes, int n_in,
                              void* d_out, int out_size, void* d_ws, size_t ws_size,
                              hipStream_t stream) {
    const float* slow  = (const float*)d_in[0];
    const float* gamma = (const float*)d_in[1];
    const float* beta  = (const float*)d_in[2];
    const float* W1    = (const float*)d_in[3];
    const float* b1    = (const float*)d_in[4];
    const float* W2    = (const float*)d_in[5];
    const float* b2    = (const float*)d_in[6];
    float* out = (float*)d_out;

    const int Btot = in_sizes[0] / D_MODEL;   // 16384

    // workspace layout (bytes)
    char* ws = (char*)d_ws;
    _Float16* B1pack = (_Float16*)ws;                                  // 2,097,152
    _Float16* B2pack = (_Float16*)(ws + 2097152);                      //    98,304
    float*    s1     = (float*)(ws + 2097152 + 98304);                 //     2,048
    float*    b1p    = (float*)(ws + 2097152 + 98304 + 2048);          //     2,048

    pack_w1<<<(D_MODEL * HIDDEN) / 256, 256, 0, stream>>>(W1, gamma, B1pack);
    pack_w2<<<(HIDDEN * NPARAM) / 256, 256, 0, stream>>>(W2, B2pack);
    col_reduce<<<HIDDEN, 256, 0, stream>>>(W1, gamma, beta, b1, s1, b1p);

    hope_main<<<Btot / MTILE, 512, SMEM_BYTES, stream>>>(
        slow, B1pack, B2pack, s1, b1p, b2, out, Btot);
}
